// YatPerformerTensorCausalAttention_84241488543838
// MI455X (gfx1250) — matrix operand order repaired and checked by reading, not compiled, not run on hardware
//
#include <hip/hip_runtime.h>
#include <hip/hip_bf16.h>

typedef _Float16 f16;
typedef __attribute__((ext_vector_type(16))) _Float16 v16h;
typedef __attribute__((ext_vector_type(8)))  float    v8f;
typedef __attribute__((ext_vector_type(4)))  int      v4i;
typedef __attribute__((ext_vector_type(8)))  int      v8i;

// ---------------------------------------------------------------------------
// WMMA helpers (gfx1250: V_WMMA_F32_16X16X32_F16, wave32)
//   A 16x32 f16: lane m=L&15, half=L>>4; elem e -> K=e+16*half
//   B 32x16 f16 (stored as N x K, K contiguous): lane n=L&15; elem e -> K=e+16*half
//   C 16x16 f32: lane n=L&15; elem r -> m = r + 8*half
// ---------------------------------------------------------------------------
__device__ __forceinline__ v8f wmma(v16h a, v16h b, v8f c) {
  return __builtin_amdgcn_wmma_f32_16x16x32_f16(false, a, false, b, (short)0, c,
                                                false, false);
}

__device__ __forceinline__ v16h frag_a_g(const f16* __restrict__ A, int ld,
                                         int row0, int k0, int lane) {
  const int m = lane & 15, hf = lane >> 4;
  const f16* p = A + (size_t)(row0 + m) * ld + k0;
  v16h a;
#pragma unroll
  for (int e = 0; e < 16; ++e) {
    const int kk = e + (hf << 4);
    a[e] = p[kk];
  }
  return a;
}

__device__ __forceinline__ v16h frag_b_g(const f16* __restrict__ Bt, int ld,
                                         int n0, int k0, int lane) {
  const int n = lane & 15, hf = lane >> 4;
  const f16* p = Bt + (size_t)(n0 + n) * ld + k0 + (hf << 4);
  v16h b;
#pragma unroll
  for (int e = 0; e < 16; ++e) b[e] = p[e];
  return b;
}

__device__ __forceinline__ v16h frag_b_lds_h(const f16* B, int ld, int n0,
                                             int k0, int lane) {
  const int n = lane & 15, hf = lane >> 4;
  const f16* p = B + (n0 + n) * ld + k0 + (hf << 4);
  v16h b;
#pragma unroll
  for (int e = 0; e < 16; ++e) b[e] = p[e];
  return b;
}

__device__ __forceinline__ v16h frag_a_lds_f32(const float* S, int ld,
                                               int row0, int k0, int lane) {
  const int m = lane & 15, hf = lane >> 4;
  const float* p = S + (row0 + m) * ld + k0;
  v16h a;
#pragma unroll
  for (int e = 0; e < 16; ++e) {
    const int kk = e + (hf << 4);
    a[e] = (f16)p[kk];
  }
  return a;
}

// ---------------------------------------------------------------------------
// Tensor Data Mover: 2D tile load global -> LDS (D# per CDNA5 ISA).
// Loads a [64 rows (dim1, stride ld=1024 elems)] x [128 cols (dim0)] f16 tile,
// padding LDS with 1 DWORD every 64 DWORDs => LDS row stride 130 f16.
// ---------------------------------------------------------------------------
__device__ __forceinline__ void tdm_load_vt(uint32_t lds_off, const f16* gsrc) {
  const uint64_t ga = (uint64_t)(uintptr_t)gsrc;
  v4i g0;
  g0[0] = 1;                                    // count=1 (valid user D#)
  g0[1] = (int)lds_off;                         // lds_addr (bytes)
  g0[2] = (int)(uint32_t)ga;                    // global_addr[31:0]
  g0[3] = (int)(((uint32_t)(ga >> 32) & 0x01FFFFFFu) | (2u << 30)); // addr hi | type=2
  v8i g1;
  g1[0] = (int)((1u << 16) | (1u << 20) | (5u << 22)); // data_size=2B, pad_en, pad_interval=64dw, pad=1dw
  g1[1] = (int)(1024u << 16);                   // tensor_dim0 = 1024
  g1[2] = (int)(64u << 16);                     // tensor_dim1 = 64
  g1[3] = (int)(128u << 16);                    // tile_dim0 = 128
  g1[4] = 64;                                   // tile_dim1 = 64
  g1[5] = 1024;                                 // tensor_dim0_stride = 1024
  g1[6] = (int)(1024u << 16);                   // tensor_dim1_stride = 1024 (lo16)
  g1[7] = 0;
  asm volatile("tensor_load_to_lds %0, %1" : : "s"(g0), "s"(g1) : "memory");
}

// ---------------------------------------------------------------------------
// f32 -> f16 convert
// ---------------------------------------------------------------------------
__global__ __launch_bounds__(256) void cvt_f16(const float* __restrict__ src,
                                               f16* __restrict__ dst, int n) {
  int i = blockIdx.x * 256 + threadIdx.x;
  if (i < n) dst[i] = (f16)src[i];
}

// ---------------------------------------------------------------------------
// C(MxN,f32) = A(MxK,f16 row-major) * Bt(NxK,f16 row-major)^T
// block 256 = 8 waves, 128x128 tile / block, wave tile 32x64
// ---------------------------------------------------------------------------
__global__ __launch_bounds__(256) void gemm_f16(const f16* __restrict__ A,
                                                const f16* __restrict__ Bt,
                                                float* __restrict__ C,
                                                int K, int ldc) {
  const int lane = threadIdx.x & 31, wid = threadIdx.x >> 5;
  const int wm = wid >> 1, wn = wid & 1;
  const int mb = blockIdx.x * 128 + wm * 32;
  const int nb = blockIdx.y * 128 + wn * 64;
  v8f zero = {};
  v8f acc[2][4];
#pragma unroll
  for (int i = 0; i < 2; ++i)
#pragma unroll
    for (int j = 0; j < 4; ++j) acc[i][j] = zero;

  for (int k0 = 0; k0 < K; k0 += 32) {
    if (k0 + 64 < K) {  // warm GL2/L0 two k-steps ahead (global_prefetch_b8)
      __builtin_prefetch(A + (size_t)(mb + (lane & 15)) * K + k0 + 64, 0, 3);
      __builtin_prefetch(Bt + (size_t)(nb + (lane & 15)) * K + k0 + 64, 0, 3);
    }
    v16h a0 = frag_a_g(A, K, mb, k0, lane);
    v16h a1 = frag_a_g(A, K, mb + 16, k0, lane);
#pragma unroll
    for (int j = 0; j < 4; ++j) {
      v16h b = frag_b_g(Bt, K, nb + 16 * j, k0, lane);
      acc[0][j] = wmma(a0, b, acc[0][j]);
      acc[1][j] = wmma(a1, b, acc[1][j]);
    }
  }
  const int n = lane & 15, hf = lane >> 4;
#pragma unroll
  for (int i = 0; i < 2; ++i)
#pragma unroll
    for (int j = 0; j < 4; ++j)
#pragma unroll
      for (int r = 0; r < 8; ++r)
        C[(size_t)(mb + 16 * i + r + 8 * hf) * ldc + nb + 16 * j + n] =
            acc[i][j][r];
}

// ---------------------------------------------------------------------------
// Featurize: per token-head, l2norm -> count sketch (ds_add_f32) -> circular
// conv (P=64) -> PRF exp features -> phi = poly (x) prf (512 f16).
// 4 token-heads per 256-thread block; 64 threads (one per d) per token-head.
// ---------------------------------------------------------------------------
__global__ __launch_bounds__(256) void featurize(
    const float* __restrict__ qkv, const float* __restrict__ omega,
    const float* __restrict__ qnode, const float* __restrict__ qweight,
    const int* __restrict__ h1, const int* __restrict__ s1,
    const int* __restrict__ h2, const int* __restrict__ s2,
    f16* __restrict__ phiq, f16* __restrict__ phik, f16* __restrict__ phikT,
    f16* __restrict__ vT) {
  __shared__ float an[4][64], c1[4][64], c2[4][64], red[4][64], prf[4][8];
  const int g = threadIdx.x >> 6;
  const int d = threadIdx.x & 63;
  const int id = blockIdx.x * 4 + g;               // 0 .. 16383
  const int b = id >> 13, h = (id >> 10) & 7, t = id & 1023;
  const int bh = b * 8 + h;

  const float s0 = qnode[0];
  const float w0 = qweight[0];
  const float sq2s = sqrtf(2.0f * fmaxf(s0, 0.0f));
  const float sqw  = sqrtf(fmaxf(w0, 0.0f));

  const size_t rowbase = (size_t)(b * 1024 + t) * 1536 + h * 64;
  // v, transposed: vT[bh][d][t]
  vT[((size_t)bh * 64 + d) * 1024 + t] = (f16)qkv[rowbase + 1024 + d];

  const int hh1 = h1[d], hh2 = h2[d];
  const float ss1 = (float)s1[d], ss2 = (float)s2[d];

  for (int which = 0; which < 2; ++which) {  // 0 = q, 1 = k
    const float av = qkv[rowbase + which * 512 + d];
    red[g][d] = av * av;
    __syncthreads();
#pragma unroll
    for (int s = 32; s > 0; s >>= 1) {
      if (d < s) red[g][d] += red[g][d + s];
      __syncthreads();
    }
    const float a = av / fmaxf(sqrtf(red[g][0]), 1e-12f);
    an[g][d] = a;
    c1[g][d] = 0.0f;
    c2[g][d] = 0.0f;
    __syncthreads();
    atomicAdd(&c1[g][hh1], a * ss1);   // ds_add_f32
    atomicAdd(&c2[g][hh2], a * ss2);
    if (d < 8) {
      float pr = 0.0f;
      for (int dd = 0; dd < 64; ++dd) pr += an[g][dd] * omega[(h * 64 + dd) * 8 + d];
      float z = fminf(fmaxf(pr * sq2s - s0, -20.0f), 20.0f);
      prf[g][d] = expf(z) * 0.35355339059327373f * sqw;  // /sqrt(M) * sqrt(w)
    }
    __syncthreads();
    // circular convolution: ts[d] = sum_i c1[i]*c2[(d-i)&63]; /sqrt(P)
    float ts = 0.0f;
    for (int i = 0; i < 64; ++i) ts += c1[g][i] * c2[g][(d - i) & 63];
    ts *= 0.125f;
    f16* dst = (which == 0 ? phiq : phik) + ((size_t)bh * 1024 + t) * 512 + d * 8;
#pragma unroll
    for (int m = 0; m < 8; ++m) {
      float val = ts * prf[g][m];
      dst[m] = (f16)val;
      if (which == 1)
        phikT[((size_t)bh * 512 + d * 8 + m) * 1024 + t] = (f16)val;
    }
    __syncthreads();
  }
}

// ---------------------------------------------------------------------------
// Chunked causal linear attention, one workgroup (8 waves) per (b,h).
// KV state (512x64 f32) lives in WMMA C-layout registers: wave w owns
// f-rows [64w, 64w+64) as st[4][4] v8f.  S and ctx partials in LDS.
// V^T chunk tile staged into LDS by the Tensor Data Mover each chunk.
// ---------------------------------------------------------------------------
__global__ __launch_bounds__(256) void chunk_attn(
    const f16* __restrict__ phiq, const f16* __restrict__ phik,
    const f16* __restrict__ phikT, const f16* __restrict__ vT,
    f16* __restrict__ attn) {
  __shared__ float S[128][132];     // masked scores (f32)
  __shared__ float ctx[128][66];    // cross-wave ctx_inter reduction
  __shared__ float kst[512];        // k_state
  __shared__ float den[128];
  __shared__ f16   Vt[64][130];     // TDM-staged V^T tile (pad 1 dword / row)

  const int bh = blockIdx.x;
  const int tid = threadIdx.x;
  const int lane = tid & 31, wid = tid >> 5;
  const int n = lane & 15, hf = lane >> 4;

  const f16* Q  = phiq  + (size_t)bh * 1024 * 512;
  const f16* Kc = phik  + (size_t)bh * 1024 * 512;
  const f16* KT = phikT + (size_t)bh * 512 * 1024;
  const f16* V  = vT    + (size_t)bh * 64 * 1024;

  v8f zero = {};
  v8f st[4][4];                     // KV state: rows 64*wid+16i, cols 16j
#pragma unroll
  for (int i = 0; i < 4; ++i)
#pragma unroll
    for (int j = 0; j < 4; ++j) st[i][j] = zero;
  for (int i = tid; i < 512; i += 256) kst[i] = 0.0f;
  __syncthreads();

  for (int ci = 0; ci < 8; ++ci) {
    const int t0 = ci * 128;

    // async: TDM pulls this chunk's V^T (64x128 f16) into LDS while we
    // compute P1/P2/P3a; consumed in P3b/P4 after s_wait_tensorcnt+barrier.
    if (wid == 0)
      tdm_load_vt((uint32_t)(uintptr_t)&Vt[0][0], V + t0);

    // -------- P1: S = Qphi * Kphi^T, causal-masked (wave w: rows 16w..) ----
#pragma unroll
    for (int jb = 0; jb < 8; jb += 4) {
      v8f acc[4];
#pragma unroll
      for (int j = 0; j < 4; ++j) acc[j] = zero;
      for (int k0 = 0; k0 < 512; k0 += 32) {
        v16h a = frag_a_g(Q, 512, t0 + wid * 16, k0, lane);
#pragma unroll
        for (int j = 0; j < 4; ++j) {
          v16h b = frag_b_g(Kc, 512, t0 + (jb + j) * 16, k0, lane);
          acc[j] = wmma(a, b, acc[j]);
        }
      }
#pragma unroll
      for (int j = 0; j < 4; ++j)
#pragma unroll
        for (int r = 0; r < 8; ++r) {
          int ml = wid * 16 + r + 8 * hf;
          int nl = (jb + j) * 16 + n;
          S[ml][nl] = (nl <= ml) ? acc[j][r] : 0.0f;
        }
    }
    __syncthreads();

    // -------- P2: den[t] = max(Qphi[t].kst + rowsum(S[t]), eps); zero ctx --
    if (tid < 128) {
      const f16* qr = Q + (size_t)(t0 + tid) * 512;
      float dsum = 0.0f;
      for (int f = 0; f < 512; ++f) dsum += (float)qr[f] * kst[f];
      float rs = 0.0f;
      for (int j = 0; j < 128; ++j) rs += S[tid][j];
      den[tid] = fmaxf(dsum + rs, 1e-6f);
    } else {
      for (int idx = tid - 128; idx < 128 * 64; idx += 128)
        ctx[idx >> 6][idx & 63] = 0.0f;
    }
    __syncthreads();

    // -------- P3a: ctx_inter partials: Qphi[:, f_w] * KV_w (registers) -----
#pragma unroll 1
    for (int mi = 0; mi < 8; ++mi) {
      v8f acc[4];
#pragma unroll
      for (int j = 0; j < 4; ++j) acc[j] = zero;
#pragma unroll
      for (int kb = 0; kb < 2; ++kb) {           // K = 64 per wave
        const int i0 = kb * 2;
        v16h a = frag_a_g(Q, 512, t0 + mi * 16, wid * 64 + kb * 32, lane);
#pragma unroll
        for (int j = 0; j < 4; ++j) {
          v16h b;
#pragma unroll
          for (int e = 0; e < 16; ++e) {
            const int src = (lane & 15) | ((e >= 8) ? 16 : 0);
            float v0 = __shfl(st[i0][j][e & 7], src, 32);
            float v1 = __shfl(st[i0 + 1][j][e & 7], src, 32);
            b[e] = (f16)(hf ? v1 : v0);
          }
          acc[j] = wmma(a, b, acc[j]);
        }
      }
#pragma unroll
      for (int j = 0; j < 4; ++j)
#pragma unroll
        for (int r = 0; r < 8; ++r)
          atomicAdd(&ctx[mi * 16 + r + 8 * hf][j * 16 + n], acc[j][r]);
    }
    if (wid == 0) __builtin_amdgcn_s_wait_tensorcnt(0);  // V^T tile landed
    __syncthreads();

    // -------- P3b: ctx_intra = S * V ; out = (intra + inter)/den -----------
    {
      v8f acc[4];
#pragma unroll
      for (int j = 0; j < 4; ++j) acc[j] = zero;
#pragma unroll
      for (int kb = 0; kb < 4; ++kb) {
        v16h a = frag_a_lds_f32(&S[0][0], 132, wid * 16, kb * 32, lane);
#pragma unroll
        for (int j = 0; j < 4; ++j) {
          v16h b = frag_b_lds_h(&Vt[0][0], 130, j * 16, kb * 32, lane);
          acc[j] = wmma(a, b, acc[j]);
        }
      }
      const int b_ = bh >> 3, h_ = bh & 7;
#pragma unroll
      for (int j = 0; j < 4; ++j)
#pragma unroll
        for (int r = 0; r < 8; ++r) {
          const int ml = wid * 16 + r + 8 * hf;
          float val = (acc[j][r] + ctx[ml][j * 16 + n]) / den[ml];
          const size_t row = (size_t)(b_ * 1024 + t0 + ml);
          attn[row * 512 + h_ * 64 + j * 16 + n] = (f16)val;
        }
    }
    __syncthreads();

    // -------- P4: KV_w += Kphi^T * V (pure register accumulation) ----------
#pragma unroll
    for (int kb = 0; kb < 4; ++kb) {
      v16h a[4];
#pragma unroll
      for (int i = 0; i < 4; ++i)
        a[i] = frag_a_g(KT, 1024, wid * 64 + i * 16, t0 + kb * 32, lane);
#pragma unroll
      for (int j = 0; j < 4; ++j) {
        v16h b = frag_b_lds_h(&Vt[0][0], 130, j * 16, kb * 32, lane);
#pragma unroll
        for (int i = 0; i < 4; ++i) st[i][j] = wmma(a[i], b, st[i][j]);
      }
    }

    // -------- P5: k_state += column sums of Kphi chunk ---------------------
    for (int f = tid; f < 512; f += 256) {
      float s = kst[f];
      const f16* kp = Kc + (size_t)t0 * 512 + f;
      for (int tt = 0; tt < 128; ++tt) s += (float)kp[(size_t)tt * 512];
      kst[f] = s;
    }
    __syncthreads();   // also fences Vt reads before next chunk's TDM write
  }
}

// ---------------------------------------------------------------------------
// Host launch
// ---------------------------------------------------------------------------
extern "C" void kernel_launch(void* const* d_in, const int* in_sizes, int n_in,
                              void* d_out, int out_size, void* d_ws,
                              size_t ws_size, hipStream_t stream) {
  (void)in_sizes; (void)n_in; (void)out_size; (void)ws_size;
  const float* x      = (const float*)d_in[0];
  const float* w_qkv  = (const float*)d_in[1];
  const float* w_out  = (const float*)d_in[2];
  const float* omega  = (const float*)d_in[3];
  const float* qnode  = (const float*)d_in[4];
  const float* qweig  = (const float*)d_in[5];
  const int*   h1     = (const int*)d_in[6];
  const int*   s1     = (const int*)d_in[7];
  const int*   h2     = (const int*)d_in[8];
  const int*   s2     = (const int*)d_in[9];

  char* ws = (char*)d_ws;
  float* qkv  = (float*)(ws + 0);                 // 2048*1536 f32
  f16*   xh   = (f16*)(ws + 12582912);            // 2048*512
  f16*   wqh  = (f16*)(ws + 14680064);            // 1536*512
  f16*   woh  = (f16*)(ws + 16252928);            // 512*512
  f16*   phiq = (f16*)(ws + 16777216);            // 16*1024*512
  f16*   phik = (f16*)(ws + 33554432);
  f16*   phikT= (f16*)(ws + 50331648);
  f16*   vTT  = (f16*)(ws + 67108864);            // 16*64*1024
  f16*   attn = (f16*)(ws + 69206016);            // 2048*512

  cvt_f16<<<4096, 256, 0, stream>>>(x, xh, 2048 * 512);
  cvt_f16<<<3072, 256, 0, stream>>>(w_qkv, wqh, 1536 * 512);
  cvt_f16<<<1024, 256, 0, stream>>>(w_out, woh, 512 * 512);

  // qkv = x @ w_qkv^T : M=2048 N=1536 K=512
  gemm_f16<<<dim3(16, 12), 256, 0, stream>>>(xh, wqh, qkv, 512, 1536);

  featurize<<<4096, 256, 0, stream>>>(qkv, omega, qnode, qweig, h1, s1, h2, s2,
                                      phiq, phik, phikT, vTT);

  chunk_attn<<<16, 256, 0, stream>>>(phiq, phik, phikT, vTT, attn);

  // y = attn @ w_out^T : M=2048 N=512 K=512 -> d_out (f32)
  gemm_f16<<<dim3(16, 4), 256, 0, stream>>>(attn, woh, (float*)d_out, 512, 512);
}
